// MixtureOfExperts_84439057039463
// MI455X (gfx1250) — matrix operand order repaired and checked by reading, not compiled, not run on hardware
//
#include <hip/hip_runtime.h>

#define V_SZ 16000
#define D_SZ 1024
#define H_SZ 1024
#define E_SZ 8
#define C_SZ 16
#define K_TOP 2
#define B_SZ 64
#define S_SZ 512
#define GATE_H 256

typedef __attribute__((ext_vector_type(16))) __bf16        v16bf;
typedef __attribute__((ext_vector_type(8)))  float         v8f;
typedef __attribute__((ext_vector_type(4)))  float         v4f;
typedef __attribute__((ext_vector_type(4)))  unsigned int  v4u;

union Frag {
    v16bf v;
    __bf16 h[16];
    unsigned int u[8];
    v4u q[2];
};

union Pack2 {
    __bf16 h[2];
    unsigned int u;
};

union Pack8 {
    __bf16 h[8];
    v4u q;
};

// ---------------------------------------------------------------------------
// Kernel 0: pre-pack W1 (f32 row-major [E,D,H]) into fragment-native bf16.
// Block(e, nt16, ks) of 1KB = 32 lanes x 8 uints = one wave-wide WMMA
// B-operand image for 16-col tile nt16, K-step ks. GEMM loads B as two
// coalesced global_load_b128 per fragment.
// ---------------------------------------------------------------------------
__global__ __launch_bounds__(256) void pack_w1_kernel(
    const float* __restrict__ exp_w1, unsigned int* __restrict__ w1p)
{
    const int t    = blockIdx.x * 256 + threadIdx.x; // (e,nt,ks,lane)
    const int lane = t & 31;
    const int ks   = (t >> 5) & 31;
    const int nt   = (t >> 10) & 63;
    const int e    = t >> 16;

    const int n  = nt * 16 + (lane & 15);
    const int kg = ks * 32 + ((lane >> 4) << 4);   // B lane-half K base

    const float* __restrict__ w1 = exp_w1 + (size_t)e * D_SZ * H_SZ;
    unsigned int* __restrict__ dst =
        w1p + (((size_t)(e * 64 + nt) * 32 + ks) << 8) + lane * 8;

    #pragma unroll
    for (int j = 0; j < 8; ++j) {
        const int kr = kg + 2 * j;
        Pack2 pk;
        pk.h[0] = (__bf16)w1[(size_t)kr * H_SZ + n];
        pk.h[1] = (__bf16)w1[(size_t)(kr + 1) * H_SZ + n];
        dst[j] = pk.u;
    }
}

// ---------------------------------------------------------------------------
// Kernel 1: pooled embedding mean + gate MLP + softmax + top-2 routing.
// One workgroup (256 threads) per batch row.
// ---------------------------------------------------------------------------
__global__ __launch_bounds__(256) void gate_kernel(
    const int* __restrict__ x, const float* __restrict__ emb,
    const float* __restrict__ gw1, const float* __restrict__ gb1,
    const float* __restrict__ gw2, const float* __restrict__ gb2,
    int* __restrict__ idx_ws, float* __restrict__ rw_ws)
{
    __shared__ int   ids[S_SZ];
    __shared__ float pooled[D_SZ];
    __shared__ float g1[GATE_H];
    __shared__ float logit[E_SZ];

    const int b = blockIdx.x;
    const int tid = threadIdx.x;

    for (int i = tid; i < S_SZ; i += 256) ids[i] = x[b * S_SZ + i];
    __syncthreads();

    for (int d = tid; d < D_SZ; d += 256) {
        float acc = 0.f;
        for (int s = 0; s < S_SZ; ++s)
            acc += emb[(size_t)ids[s] * D_SZ + d];
        pooled[d] = acc * (1.0f / S_SZ);
    }
    __syncthreads();

    {
        float acc = gb1[tid];
        for (int d = 0; d < D_SZ; ++d)
            acc += pooled[d] * gw1[d * GATE_H + tid];
        g1[tid] = fmaxf(acc, 0.f);
    }
    __syncthreads();

    if (tid < E_SZ) {
        float acc = gb2[tid];
        for (int j = 0; j < GATE_H; ++j)
            acc += g1[j] * gw2[j * E_SZ + tid];
        logit[tid] = acc;
    }
    __syncthreads();

    if (tid == 0) {
        float mx = logit[0];
        for (int e = 1; e < E_SZ; ++e) mx = fmaxf(mx, logit[e]);
        float p[E_SZ];
        for (int e = 0; e < E_SZ; ++e) p[e] = __expf(logit[e] - mx);
        int i0 = 0;
        for (int e = 1; e < E_SZ; ++e) if (p[e] > p[i0]) i0 = e;
        int i1 = (i0 == 0) ? 1 : 0;
        for (int e = 0; e < E_SZ; ++e)
            if (e != i0 && p[e] > p[i1]) i1 = e;
        float w0 = p[i0], w1 = p[i1], t = w0 + w1;
        idx_ws[b * K_TOP + 0] = i0;
        idx_ws[b * K_TOP + 1] = i1;
        rw_ws[b * K_TOP + 0] = w0 / t;
        rw_ws[b * K_TOP + 1] = w1 / t;
    }
}

// ---------------------------------------------------------------------------
// Kernel 2: fused expert GEMM:  p[b,k,:] = mean_s relu(tok @ W1[e] + b1[e])
// Grid (4 n-tiles of 256 cols, 128 (b,k) pairs), 8 wave32 per WG.
// M-chunk = 64 token rows staged in LDS: each K-step loads two B fragments
// (4x coalesced global_load_b128 from the packed L2-resident weights) and
// reuses them across 4 M-tiles -> 8 WMMAs per K-step. Two LDS base pointers
// (rows 0-31 / rows 32-63) keep all ds_load offsets within the 16-bit DS
// offset field, so the inner loop has no per-iteration address VALU (and no
// WMMA WAR hazard nops). K-loop rolled to keep ~120 VGPRs, no spills.
// Epilogue fuses bias+relu+column-mean via ds_add_f32.
// ---------------------------------------------------------------------------
#define ASTRIDE 1032   // bf16 elems per LDS row: 2064B rows, 16B aligned,
                       // b128 reads from 16 rows cover all 64 banks once.
#define MT 4           // M-tiles (16 rows each) per chunk

__global__ __launch_bounds__(256) void expert_kernel(
    const int* __restrict__ x, const float* __restrict__ exp_emb,
    const unsigned int* __restrict__ w1p, const float* __restrict__ exp_b1,
    const int* __restrict__ idx_ws, float* __restrict__ p_ws)
{
    __shared__ __bf16 Alds[MT * 16 * ASTRIDE];   // 132 KB
    __shared__ float  p_lds[256];
    __shared__ int    ids[S_SZ];

    const int tid   = threadIdx.x;
    const int ntile = blockIdx.x;        // 0..3   (256 columns of H each)
    const int bk    = blockIdx.y;        // 0..127 (b * K_TOP + k)
    const int b     = bk >> 1;

    const int e    = idx_ws[bk];
    const int wave = tid >> 5;
    const int lane = tid & 31;
    const int nl   = lane & 15;
    const int kA   = (lane >> 4) << 4;   // A lane-half: K base 0 / 16

    const int nt16a = ntile * 16 + wave;          // 16-col tile ids
    const int nt16b = nt16a + 8;
    const int n_a   = nt16a * 16 + nl;            // global H columns
    const int n_b   = nt16b * 16 + nl;

    for (int i = tid; i < S_SZ; i += 256) ids[i] = x[b * S_SZ + i];
    p_lds[tid] = 0.f;
    __syncthreads();

    const size_t ebase = (size_t)e * V_SZ * D_SZ;
    const unsigned int* __restrict__ w1e = w1p + ((size_t)e << 19); // e*64*32*256
    const unsigned int* __restrict__ bb0 = w1e + (((size_t)nt16a * 32) << 8) + lane * 8;
    const unsigned int* __restrict__ bb1 = w1e + (((size_t)nt16b * 32) << 8) + lane * 8;
    const float bias0 = exp_b1[e * H_SZ + n_a];
    const float bias1 = exp_b1[e * H_SZ + n_b];

    // Two stable A bases: rows 0-31 and rows 32-63. All mt offsets from these
    // bases (0 / 33024 bytes) fit the 16-bit DS immediate offset field.
    const unsigned int* __restrict__ arow01 =
        (const unsigned int*)(&Alds[nl * ASTRIDE]);
    const unsigned int* __restrict__ arow23 =
        (const unsigned int*)(&Alds[(32 + nl) * ASTRIDE]);
    const int MOFF = 16 * (ASTRIDE / 2);   // 16 rows in uints (=33024 bytes)

    for (int mc = 0; mc < S_SZ / (MT * 16); ++mc) {
        // ---- stage A: gather 64 token rows, f32 -> bf16, vectorized ----
        // thread handles 8 consecutive cols: 2x b128 load, 1x ds_store_b128
        #pragma unroll 2
        for (int i = 0; i < (MT * 16 * D_SZ) / (256 * 8); ++i) {   // 32 iters
            int f8 = i * 256 + tid;
            int r  = f8 >> 7;                 // row within chunk (8-elem groups/row=128)
            int c  = (f8 & 127) << 3;         // column (multiple of 8)
            const float* src =
                exp_emb + ebase + (size_t)ids[mc * (MT * 16) + r] * D_SZ + c;
            v4f x0 = *(const v4f*)(src);
            v4f x1 = *(const v4f*)(src + 4);
            Pack8 pk;
            #pragma unroll
            for (int j = 0; j < 4; ++j) {
                pk.h[j]     = (__bf16)x0[j];
                pk.h[4 + j] = (__bf16)x1[j];
            }
            *(v4u*)(&Alds[r * ASTRIDE + c]) = pk.q;
        }
        __syncthreads();

        v8f acc[MT][2] = {};

        #pragma unroll 1
        for (int ks = 0; ks < D_SZ / 32; ++ks) {
            Frag b0, b1;
            b0.q[0] = *(const v4u*)(bb0 + (ks << 8));
            b0.q[1] = *(const v4u*)(bb0 + (ks << 8) + 4);
            b1.q[0] = *(const v4u*)(bb1 + (ks << 8));
            b1.q[1] = *(const v4u*)(bb1 + (ks << 8) + 4);
            __builtin_prefetch(bb0 + ((ks + 1) << 8), 0, 1);
            __builtin_prefetch(bb1 + ((ks + 1) << 8), 0, 1);
            const int aoff = (ks << 4) + (kA >> 1);
            #pragma unroll
            for (int mt = 0; mt < MT; ++mt) {
                const unsigned int* base = (mt < 2) ? arow01 : arow23;
                const int moff = (mt & 1) * MOFF;
                Frag a;
                a.q[0] = *(const v4u*)(base + moff + aoff);
                a.q[1] = *(const v4u*)(base + moff + aoff + 4);
                acc[mt][0] = __builtin_amdgcn_wmma_f32_16x16x32_bf16(
                    false, a.v, false, b0.v, (short)0, acc[mt][0], false, false);
                acc[mt][1] = __builtin_amdgcn_wmma_f32_16x16x32_bf16(
                    false, a.v, false, b1.v, (short)0, acc[mt][1], false, false);
            }
        }

        // ---- epilogue: bias + relu + column sum over the 64 rows ----
        float s0 = 0.f, s1 = 0.f;
        #pragma unroll
        for (int mt = 0; mt < MT; ++mt) {
            #pragma unroll
            for (int j = 0; j < 8; ++j) {
                s0 += fmaxf(acc[mt][0][j] + bias0, 0.f);
                s1 += fmaxf(acc[mt][1][j] + bias1, 0.f);
            }
        }
        atomicAdd(&p_lds[wave * 16 + nl], s0);        // lane halves share col
        atomicAdd(&p_lds[128 + wave * 16 + nl], s1);
        __syncthreads();
    }

    p_ws[(size_t)bk * H_SZ + ntile * 256 + tid] = p_lds[tid] * (1.0f / S_SZ);
}

// ---------------------------------------------------------------------------
// Kernel 3: out[b,c] = sum_k rw[b,k] * (p[b,k,:] @ W2[e] + b2[e])[c]
// ---------------------------------------------------------------------------
__global__ __launch_bounds__(256) void out_kernel(
    const float* __restrict__ p_ws, const int* __restrict__ idx_ws,
    const float* __restrict__ rw_ws, const float* __restrict__ exp_w2,
    const float* __restrict__ exp_b2, float* __restrict__ out)
{
    __shared__ float part[256];
    const int b = blockIdx.x;
    const int tid = threadIdx.x;
    const int c = tid & 15;
    const int g = tid >> 4;

    float result = 0.f;
    for (int k = 0; k < K_TOP; ++k) {
        const int e = idx_ws[b * K_TOP + k];
        const float rw = rw_ws[b * K_TOP + k];
        const float* p  = p_ws + (size_t)(b * K_TOP + k) * H_SZ;
        const float* w2 = exp_w2 + (size_t)e * H_SZ * C_SZ;
        float acc = 0.f;
        for (int h = g; h < H_SZ; h += 16)
            acc += p[h] * w2[h * C_SZ + c];
        part[tid] = acc;
        __syncthreads();
        if (g == 0) {
            float v = 0.f;
            for (int gg = 0; gg < 16; ++gg) v += part[gg * 16 + c];
            result += rw * (v + exp_b2[e * C_SZ + c]);
        }
        __syncthreads();
    }
    if (g == 0) out[b * C_SZ + c] = result;
}

// ---------------------------------------------------------------------------
extern "C" void kernel_launch(void* const* d_in, const int* in_sizes, int n_in,
                              void* d_out, int out_size, void* d_ws, size_t ws_size,
                              hipStream_t stream) {
    const int*   x       = (const int*)  d_in[0];
    const float* emb     = (const float*)d_in[1];
    const float* gate_w1 = (const float*)d_in[2];
    const float* gate_b1 = (const float*)d_in[3];
    const float* gate_w2 = (const float*)d_in[4];
    const float* gate_b2 = (const float*)d_in[5];
    const float* exp_emb = (const float*)d_in[6];
    const float* exp_w1  = (const float*)d_in[7];
    const float* exp_b1  = (const float*)d_in[8];
    const float* exp_w2  = (const float*)d_in[9];
    const float* exp_b2  = (const float*)d_in[10];
    float* out = (float*)d_out;

    char* ws = (char*)d_ws;
    int*          idx_ws = (int*)ws;                     // B*K ints   (512 B)
    float*        rw_ws  = (float*)(ws + 512);           // B*K floats (512 B)
    float*        p_ws   = (float*)(ws + 1024);          // B*K*H f32  (512 KB)
    unsigned int* w1p    = (unsigned int*)(ws + (1 << 20)); // E*D*H bf16 (16 MB)

    pack_w1_kernel<<<(E_SZ * 64 * 32 * 32) / 256, 256, 0, stream>>>(exp_w1, w1p);
    gate_kernel<<<B_SZ, 256, 0, stream>>>(x, emb, gate_w1, gate_b1,
                                          gate_w2, gate_b2, idx_ws, rw_ws);
    expert_kernel<<<dim3(H_SZ / 256, B_SZ * K_TOP), 256, 0, stream>>>(
        x, exp_emb, w1p, exp_b1, idx_ws, p_ws);
    out_kernel<<<B_SZ, 256, 0, stream>>>(p_ws, idx_ws, rw_ws,
                                         exp_w2, exp_b2, out);
}
